// StackedDecoder_49194555408960
// MI455X (gfx1250) — matrix operand order repaired and checked by reading, not compiled, not run on hardware
//
#include <hip/hip_runtime.h>
#include <stdint.h>

// ---------------------------------------------------------------------------
// CDNA5 (gfx1250) graph-LSTM. Matrix ops via v_wmma_f32_16x16x32_bf16.
// Main GEMM: one wave computes a 16x64 strip (4 tiles), reusing the A
// fragment across 4 WMMAs per k-chunk.
// ---------------------------------------------------------------------------

typedef __attribute__((ext_vector_type(16))) __bf16 v16bf;
typedef __attribute__((ext_vector_type(8)))  float  v8f;

union FragBF { uint4 q[2]; v16bf v; uint32_t u[8]; };

__device__ __forceinline__ uint32_t pack_bf16x2(float a, float b) {
  uint32_t ua = __float_as_uint(a);
  uint32_t ub = __float_as_uint(b);
  ua = (ua + 0x7FFFu + ((ua >> 16) & 1u)) >> 16;   // RNE f32 -> bf16
  ub = (ub + 0x7FFFu + ((ub >> 16) & 1u)) >> 16;
  return (ua & 0xFFFFu) | (ub << 16);
}

__device__ __forceinline__ float sigm(float x) { return 1.0f / (1.0f + __expf(-x)); }

// -------------------------------- utility ----------------------------------
__global__ void k_zero(float* __restrict__ p, long long n) {
  long long i = (long long)blockIdx.x * blockDim.x + threadIdx.x;
  if (i < n) p[i] = 0.0f;
}

__global__ void k_deg(const int* __restrict__ dst, float* __restrict__ deg, int E) {
  int i = blockIdx.x * blockDim.x + threadIdx.x;
  if (i < E) atomicAdd(&deg[dst[i]], 1.0f);
}

__global__ void k_inv(const float* __restrict__ deg, float* __restrict__ dinv, int N) {
  int i = blockIdx.x * blockDim.x + threadIdx.x;
  if (i < N) { float d = deg[i]; dinv[i] = d > 0.0f ? 1.0f / d : 0.0f; }
}

// scatter-add feat[src] into agg[dst]; 1 thread = 1 edge x 4 features
__global__ void k_scatter(const float* __restrict__ feat, const int* __restrict__ src,
                          const int* __restrict__ dst, float* __restrict__ agg, int E) {
  long long i = (long long)blockIdx.x * blockDim.x + threadIdx.x;
  long long tot = (long long)E * 16;
  if (i >= tot) return;
  int e  = (int)(i >> 4);
  int f4 = (int)(i & 15) << 2;
  int s = src[e], d = dst[e];
  float4 v = *(const float4*)(feat + (size_t)s * 64 + f4);
  float* o = agg + (size_t)d * 64 + f4;
  atomicAdd(o + 0, v.x); atomicAdd(o + 1, v.y);
  atomicAdd(o + 2, v.z); atomicAdd(o + 3, v.w);
}

// ---------------------- pack A = [inp | x_agg | h | h_agg] -----------------
// bf16, laid out exactly as the WMMA 16-bit A-matrix per-lane fragment:
// dword index = ((rt*8 + kc)*32 + lane)*8 + v
// lane: M = rt*16 + (lane&15); k0 = kc*32 + (v&4?16:0) + (lane>>4)*8 + (v&3)*2
__global__ void k_packA(const float* __restrict__ inp, const float* __restrict__ aggX,
                        const float* __restrict__ hbuf, const float* __restrict__ aggH,
                        const float* __restrict__ dinv, uint32_t* __restrict__ Apack, int N) {
  long long i = (long long)blockIdx.x * blockDim.x + threadIdx.x;
  long long tot = (long long)N * 128;
  if (i >= tot) return;
  int v = (int)(i & 7);
  int lane = (int)((i >> 3) & 31);
  int kc = (int)((i >> 8) & 7);
  int rt = (int)(i >> 11);
  int n = rt * 16 + (lane & 15);
  int half = lane >> 4;
  int k0 = kc * 32 + ((v & 4) ? 16 : 0) + half * 8 + (v & 3) * 2;
  int blk = k0 >> 6;   // 0:inp 1:x_agg 2:h 3:h_agg
  int kk = k0 & 63;
  float sc = 1.0f;
  const float* sp;
  if (blk == 0)      { sp = inp; }
  else if (blk == 1) { sp = aggX; sc = dinv[n]; }
  else if (blk == 2) { sp = hbuf; }
  else               { sp = aggH; sc = dinv[n]; }
  float a = sp[(size_t)n * 64 + kk] * sc;
  float b = sp[(size_t)n * 64 + kk + 1] * sc;
  Apack[i] = pack_bf16x2(a, b);
}

// ------- pack B = [Wx_self; Wx_neigh; Wh_self; Wh_neigh] (256x256, bf16) ---
// dword index = (((l*16+ct)*8+kc)*32+lane)*8+v
// B fragment layout: N = ct*16 + (lane&15); k0 = kc*32 + (v&4?16:0) + (lane>>4)*8 + (v&3)*2
__global__ void k_packW(const float* __restrict__ Wxs, const float* __restrict__ Wxn,
                        const float* __restrict__ Whs, const float* __restrict__ Whn,
                        uint32_t* __restrict__ Bpack) {
  int i = blockIdx.x * blockDim.x + threadIdx.x;
  if (i >= 2 * 65536) return;
  int v = i & 7;
  int lane = (i >> 3) & 31;
  int kc = (i >> 8) & 7;
  int ct = (i >> 11) & 15;
  int l  = i >> 15;
  int col = ct * 16 + (lane & 15);
  int g = col >> 6, hh = col & 63;
  int k0 = kc * 32 + ((v & 4) ? 16 : 0) + (lane >> 4) * 8 + (v & 3) * 2;
  int blk = k0 >> 6, kk = k0 & 63;
  const float* W = (blk == 0) ? Wxs : (blk == 1) ? Wxn : (blk == 2) ? Whs : Whn;
  size_t base = (((size_t)l * 4 + g) * 64 + kk) * 64 + hh;
  Bpack[i] = pack_bf16x2(W[base], W[base + 64]);   // (kk,hh) and (kk+1,hh)
}

// Wout [64,32] -> packed bf16 fragments: dword = ((ct*2+kc)*32+lane)*8+v
__global__ void k_packWout(const float* __restrict__ Wout, uint32_t* __restrict__ Wp) {
  int i = blockIdx.x * blockDim.x + threadIdx.x;
  if (i >= 1024) return;
  int v = i & 7;
  int lane = (i >> 3) & 31;
  int kc = (i >> 8) & 1;
  int ct = (i >> 9) & 1;
  int col = ct * 16 + (lane & 15);
  int k0 = kc * 32 + ((v & 4) ? 16 : 0) + (lane >> 4) * 8 + (v & 3) * 2;
  Wp[i] = pack_bf16x2(Wout[(size_t)k0 * 32 + col], Wout[(size_t)(k0 + 1) * 32 + col]);
}

// ----------------------------- main GEMM -----------------------------------
// pre[N,256] = Apack[N,256] @ Bpack[256,256], bf16 in / f32 accumulate.
// One wave per 16x64 output strip: 4 accumulators; A fragment loaded once
// per k-chunk and reused by 4 WMMAs. K=256 -> 8 k-chunks -> 32 WMMAs/wave.
__global__ __launch_bounds__(128) void k_gemm(const uint32_t* __restrict__ Apack,
                                              const uint32_t* __restrict__ Bpack,
                                              float* __restrict__ pre, int totalWaves) {
  int wave = blockIdx.x * 4 + (threadIdx.x >> 5);
  if (wave >= totalWaves) return;           // wave-uniform: EXEC stays all-ones
  int lane = threadIdx.x & 31;
  int rt  = wave >> 2;
  int ctg = wave & 3;                        // column group: cols [ctg*64, ctg*64+64)
  v8f acc0 = {}, acc1 = {}, acc2 = {}, acc3 = {};
#pragma unroll
  for (int kc = 0; kc < 8; ++kc) {
    FragBF a, b0, b1, b2, b3;
    const uint4* pa = (const uint4*)(Apack + (((size_t)rt * 8 + kc) * 32 + lane) * 8);
    a.q[0] = pa[0]; a.q[1] = pa[1];
    const uint4* pb0 = (const uint4*)(Bpack + ((((size_t)ctg * 4 + 0) * 8 + kc) * 32 + lane) * 8);
    const uint4* pb1 = (const uint4*)(Bpack + ((((size_t)ctg * 4 + 1) * 8 + kc) * 32 + lane) * 8);
    const uint4* pb2 = (const uint4*)(Bpack + ((((size_t)ctg * 4 + 2) * 8 + kc) * 32 + lane) * 8);
    const uint4* pb3 = (const uint4*)(Bpack + ((((size_t)ctg * 4 + 3) * 8 + kc) * 32 + lane) * 8);
    b0.q[0] = pb0[0]; b0.q[1] = pb0[1];
    b1.q[0] = pb1[0]; b1.q[1] = pb1[1];
    b2.q[0] = pb2[0]; b2.q[1] = pb2[1];
    b3.q[0] = pb3[0]; b3.q[1] = pb3[1];
    acc0 = __builtin_amdgcn_wmma_f32_16x16x32_bf16(false, a.v, false, b0.v, (short)0, acc0, false, false);
    acc1 = __builtin_amdgcn_wmma_f32_16x16x32_bf16(false, a.v, false, b1.v, (short)0, acc1, false, false);
    acc2 = __builtin_amdgcn_wmma_f32_16x16x32_bf16(false, a.v, false, b2.v, (short)0, acc2, false, false);
    acc3 = __builtin_amdgcn_wmma_f32_16x16x32_bf16(false, a.v, false, b3.v, (short)0, acc3, false, false);
  }
  int half = lane >> 4;
  int cnb  = ctg * 64 + (lane & 15);
  size_t rowBase = (size_t)rt * 16 + half * 8;
#pragma unroll
  for (int r = 0; r < 8; ++r) {
    float* row = pre + (rowBase + r) * 256 + cnb;
    row[0]  = acc0[r];
    row[16] = acc1[r];
    row[32] = acc2[r];
    row[48] = acc3[r];
  }
}

// ------------------------- output GEMM [N,64]@[64,32] ----------------------
__global__ __launch_bounds__(128) void k_outgemm(const float* __restrict__ h1,
                                                 const uint32_t* __restrict__ Wp,
                                                 const float* __restrict__ bout,
                                                 float* __restrict__ out, int totalTiles) {
  int wave = blockIdx.x * 4 + (threadIdx.x >> 5);
  if (wave >= totalTiles) return;
  int lane = threadIdx.x & 31;
  int rt = wave >> 1, ct = wave & 1;
  int m = lane & 15, half = lane >> 4;
  v8f acc = {};
#pragma unroll
  for (int kc = 0; kc < 2; ++kc) {
    FragBF a, b;
#pragma unroll
    for (int v = 0; v < 8; ++v) {   // convert fp32 h -> bf16 A fragment inline
      int k0 = kc * 32 + ((v & 4) ? 16 : 0) + half * 8 + (v & 3) * 2;
      const float* p = h1 + ((size_t)rt * 16 + m) * 64 + k0;
      a.u[v] = pack_bf16x2(p[0], p[1]);
    }
    const uint4* pb = (const uint4*)(Wp + (((size_t)ct * 2 + kc) * 32 + lane) * 8);
    b.q[0] = pb[0]; b.q[1] = pb[1];
    acc = __builtin_amdgcn_wmma_f32_16x16x32_bf16(false, a.v, false, b.v,
                                                  (short)0, acc, false, false);
  }
  int cn = ct * 16 + (lane & 15);
  float bb = bout[cn];
#pragma unroll
  for (int r = 0; r < 8; ++r)
    out[((size_t)rt * 16 + half * 8 + r) * 32 + cn] = acc[r] + bb;
}

// ------------------------- peephole LSTM elementwise -----------------------
__global__ void k_lstm(const float* __restrict__ pre, float* __restrict__ cbuf,
                       float* __restrict__ hbuf, const float* __restrict__ wc,
                       const float* __restrict__ bias, int N) {
  int i = blockIdx.x * blockDim.x + threadIdx.x;
  if (i >= N * 64) return;
  int n = i >> 6, hh = i & 63;
  const float* pr = pre + (size_t)n * 256;
  float c  = cbuf[i];
  float ig = sigm(pr[hh]        + wc[hh]       * c + bias[hh]);
  float fg = sigm(pr[64 + hh]   + wc[64 + hh]  * c + bias[64 + hh]);
  float ct = tanhf(pr[128 + hh]                    + bias[128 + hh]);
  float nc = fg * c + ig * ct;
  float og = sigm(pr[192 + hh]  + wc[128 + hh] * nc + bias[192 + hh]);
  cbuf[i] = nc;
  hbuf[i] = og * tanhf(nc);
}

// ---------------------------------------------------------------------------
extern "C" void kernel_launch(void* const* d_in, const int* in_sizes, int n_in,
                              void* d_out, int out_size, void* d_ws, size_t ws_size,
                              hipStream_t stream) {
  const float* x    = (const float*)d_in[0];
  const float* h0   = (const float*)d_in[1];
  const float* c0   = (const float*)d_in[2];
  const float* Wxs  = (const float*)d_in[3];
  const float* Wxn  = (const float*)d_in[4];
  const float* Whs  = (const float*)d_in[5];
  const float* Whn  = (const float*)d_in[6];
  const float* wc   = (const float*)d_in[7];
  const float* bias = (const float*)d_in[8];
  const float* Wout = (const float*)d_in[9];
  const float* bout = (const float*)d_in[10];
  const int*   src  = (const int*)d_in[11];
  const int*   dst  = (const int*)d_in[12];

  const int L = 2, F = 64, H = 64, O = 32;
  const int N = in_sizes[1] / (L * H);
  const int T = in_sizes[0] / (N * F);
  const int E = in_sizes[11];

  float* out  = (float*)d_out;
  float* hbuf = out + (size_t)T * N * O;       // h_f section, used as live h state
  float* cbuf = hbuf + (size_t)L * N * H;      // c_f section, used as live c state

  char* ws = (char*)d_ws;
  auto aln = [](size_t v) { return (v + 255) & ~(size_t)255; };
  size_t o = 0;
  float*    deginv = (float*)(ws + o);    o = aln(o + (size_t)N * 4);
  float*    aggX   = (float*)(ws + o);    o = aln(o + (size_t)N * 64 * 4);
  float*    aggH   = (float*)(ws + o);    o = aln(o + (size_t)N * 64 * 4);
  uint32_t* Apack  = (uint32_t*)(ws + o); o = aln(o + (size_t)N * 256 * 2);
  float*    pre    = (float*)(ws + o);    o = aln(o + (size_t)N * 256 * 4);
  uint32_t* Bpack  = (uint32_t*)(ws + o); o = aln(o + (size_t)L * 256 * 256 * 2);
  uint32_t* WoutP  = (uint32_t*)(ws + o); o = aln(o + 4096);
  (void)ws_size; (void)n_in; (void)out_size;

  auto nb = [](long long n, int bs) { return (unsigned)((n + bs - 1) / bs); };
  dim3 b256(256);

  // live state <- initial state
  hipMemcpyAsync(hbuf, h0, (size_t)L * N * H * 4, hipMemcpyDeviceToDevice, stream);
  hipMemcpyAsync(cbuf, c0, (size_t)L * N * H * 4, hipMemcpyDeviceToDevice, stream);

  // degree -> 1/deg (deg accumulated in aggX scratch)
  k_zero<<<nb(N, 256), b256, 0, stream>>>(aggX, N);
  k_deg<<<nb(E, 256), b256, 0, stream>>>(dst, aggX, E);
  k_inv<<<nb(N, 256), b256, 0, stream>>>(aggX, deginv, N);

  // weight packing (once per call)
  k_packW<<<nb(2 * 65536, 256), b256, 0, stream>>>(Wxs, Wxn, Whs, Whn, Bpack);
  k_packWout<<<nb(1024, 256), b256, 0, stream>>>(Wout, WoutP);

  const int rowTiles  = N / 16;                 // N = 100000 -> 6250
  const int gemmWaves = rowTiles * 4;           // 4 column groups of 64
  const int outTiles  = rowTiles * 2;           // 2 col tiles of 16

  for (int t = 0; t < T; ++t) {
    for (int j = 0; j < L; ++j) {
      const float* inp = (j == 0) ? (x + (size_t)t * N * F) : hbuf;  // layer-0 new h
      float* hj = hbuf + (size_t)j * N * H;
      float* cj = cbuf + (size_t)j * N * H;

      k_zero<<<nb((long long)2 * N * 64, 256), b256, 0, stream>>>(aggX, (long long)2 * N * 64);
      k_scatter<<<nb((long long)E * 16, 256), b256, 0, stream>>>(inp, src, dst, aggX, E);
      k_scatter<<<nb((long long)E * 16, 256), b256, 0, stream>>>(hj, src, dst, aggH, E);

      k_packA<<<nb((long long)N * 128, 256), b256, 0, stream>>>(inp, aggX, hj, aggH,
                                                                deginv, Apack, N);
      k_gemm<<<dim3((gemmWaves + 3) / 4), dim3(128), 0, stream>>>(
          Apack, Bpack + (size_t)j * 65536, pre, gemmWaves);
      k_lstm<<<nb((long long)N * 64, 256), b256, 0, stream>>>(
          pre, cj, hj, wc + (size_t)j * 3 * 64, bias + (size_t)j * 4 * 64, N);
    }
    k_outgemm<<<dim3((outTiles + 3) / 4), dim3(128), 0, stream>>>(
        hbuf + (size_t)N * H, WoutP, bout, out + (size_t)t * N * O, outTiles);
  }
}
